// TransposedAttentionTransformer_20340965114436
// MI455X (gfx1250) — matrix operand order repaired and checked by reading, not compiled, not run on hardware
//
#include <hip/hip_runtime.h>
#include <cstdint>
#include <cstddef>

// ---------------------------------------------------------------------------
// TransposedAttentionTransformer block for MI455X (gfx1250), bf16 WMMA path.
// D = A(16x32 bf16) x B(32x16 bf16) + C(16x16 f32) via v_wmma_f32_16x16x32_bf16
// ---------------------------------------------------------------------------

typedef __attribute__((ext_vector_type(16))) __bf16 v16bf;
typedef __attribute__((ext_vector_type(8)))  float  v8f;

#define DEVFN __device__ __forceinline__

DEVFN v8f wmma_bf16(v16bf a, v16bf b, v8f c) {
  // (neg_a, A, neg_b, B, c_mod, C, reuse_a, reuse_b)
  return __builtin_amdgcn_wmma_f32_16x16x32_bf16(false, a, false, b, (short)0, c,
                                                 false, false);
}

union FragU { v16bf v; uint4 q[2]; };

// A fragment (16x32, bf16) from row-major [M,K] tile origin `base`, ld elems.
// Lane l: row m = l&15, K chunks [kb..kb+7] and [kb+16..kb+23], kb=(l>>4)*8.
DEVFN v16bf load_frag_a(const __bf16* base, int ld, int lane) {
  const __bf16* p = base + (size_t)(lane & 15) * ld + ((lane >> 4) << 3);
  FragU u;
  u.q[0] = *(const uint4*)(p);
  u.q[1] = *(const uint4*)(p + 16);
  return u.v;
}

// B fragment (32x16, bf16) sourced from row-major W[N,K] (so B[k][n] = W[n][k]).
// Lane l: col n = l&15 -> W row; K chunks [kb..kb+7] and [kb+16..kb+23], kb=(l>>4)*8.
DEVFN v16bf load_frag_b(const __bf16* base, int ld, int lane) {
  const __bf16* p = base + (size_t)(lane & 15) * ld + ((lane >> 4) << 3);
  FragU u;
  u.q[0] = *(const uint4*)(p);
  u.q[1] = *(const uint4*)(p + 16);
  return u.v;
}

// ---------------------------------------------------------------------------
// Generic WMMA GEMM: C[M,N] = A[M,K](bf16) x W[N,K]^T(bf16), fused epilogues.
// Block: 256 thr = 8 waves arranged 4Mx2N; wave tile 64x32 (4x2 WMMA tiles);
// block tile 256x64.  Per K-step: 4 A-frags + 2 B-frags -> 8 WMMAs
// (1.5 b128-loads per WMMA; dual gate/up path reaches 1.0).
// EPI 0: store bf16             (qkv)
// EPI 1: f32 = acc + bias + res (proj + b_proj + shortcut)
// EPI 2: dual B/B2, bf16 = silu(acc)*acc2   (gate/up fused)
// EPI 3: f32 = acc + res        (down + residual -> output)
// ---------------------------------------------------------------------------
template<int EPI>
__global__ void __launch_bounds__(256)
gemm_bf16(const __bf16* __restrict__ A, const __bf16* __restrict__ B,
          const __bf16* __restrict__ B2, float* __restrict__ Cf,
          __bf16* __restrict__ Cb, const float* __restrict__ bias,
          const float* __restrict__ res, int N, int K)
{
  const int lane = threadIdx.x & 31;
  const int w    = threadIdx.x >> 5;
  const int m0 = blockIdx.y * 256 + (w >> 1) * 64;   // wave: 64 rows
  const int n0 = blockIdx.x * 64  + (w & 1) * 32;    // wave: 32 cols

  v8f acc[4][2]  = {};
  v8f acc2[4][2] = {};

  const __bf16* Ar[4];
  for (int im = 0; im < 4; ++im) Ar[im] = A + (size_t)(m0 + im * 16) * K;
  const __bf16* Br0 = B + (size_t)n0 * K;
  const __bf16* Br1 = B + (size_t)(n0 + 16) * K;

#pragma unroll 1
  for (int k0 = 0; k0 < K; k0 += 32) {
    if (k0 + 32 < K) {                          // hint: global_prefetch_b8
      __builtin_prefetch(Ar[0] + k0 + 32, 0, 0);
      __builtin_prefetch(Br0 + k0 + 32, 0, 0);
    }
    v16bf a[4];
    for (int im = 0; im < 4; ++im) a[im] = load_frag_a(Ar[im] + k0, K, lane);
    v16bf b0 = load_frag_b(Br0 + k0, K, lane);
    v16bf b1 = load_frag_b(Br1 + k0, K, lane);
    for (int im = 0; im < 4; ++im) {
      acc[im][0] = wmma_bf16(a[im], b0, acc[im][0]);
      acc[im][1] = wmma_bf16(a[im], b1, acc[im][1]);
    }
    if constexpr (EPI == 2) {
      v16bf c0 = load_frag_b(B2 + (size_t)n0 * K + k0, K, lane);
      v16bf c1 = load_frag_b(B2 + (size_t)(n0 + 16) * K + k0, K, lane);
      for (int im = 0; im < 4; ++im) {
        acc2[im][0] = wmma_bf16(a[im], c0, acc2[im][0]);
        acc2[im][1] = wmma_bf16(a[im], c1, acc2[im][1]);
      }
    }
  }

  // D layout: VGPR r -> row (lane>>4)*8 + r, col lane&15
  const int colt = lane & 15;
  const int rb   = (lane >> 4) << 3;
  for (int im = 0; im < 4; ++im)
    for (int in = 0; in < 2; ++in)
      for (int r = 0; r < 8; ++r) {
        const int mm = m0 + im * 16 + rb + r;
        const int nn = n0 + in * 16 + colt;
        const size_t idx = (size_t)mm * N + nn;
        const float vacc = acc[im][in][r];
        if constexpr (EPI == 0) {
          Cb[idx] = (__bf16)vacc;
        } else if constexpr (EPI == 1) {
          Cf[idx] = vacc + bias[nn] + res[idx];
        } else if constexpr (EPI == 2) {
          const float g = vacc;
          const float u = acc2[im][in][r];
          Cb[idx] = (__bf16)((g / (1.0f + __expf(-g))) * u);
        } else {
          Cf[idx] = vacc + res[idx];
        }
      }
}

// ---------------------------------------------------------------------------
// Row RMSNorm over C=1024, f32 in -> bf16 out. One 256-thread block per row.
// ---------------------------------------------------------------------------
__global__ void __launch_bounds__(256)
rmsnorm_rows(const float* __restrict__ x, const float* __restrict__ w,
             __bf16* __restrict__ out)
{
  __shared__ float red[256];
  const size_t row = blockIdx.x;
  const float4 xv = ((const float4*)(x + row * 1024))[threadIdx.x];
  red[threadIdx.x] = xv.x*xv.x + xv.y*xv.y + xv.z*xv.z + xv.w*xv.w;
  __syncthreads();
  for (int s = 128; s > 0; s >>= 1) {
    if (threadIdx.x < s) red[threadIdx.x] += red[threadIdx.x + s];
    __syncthreads();
  }
  const float rinv = rsqrtf(red[0] * (1.0f / 1024.0f) + 1e-5f);
  const float4 wv = ((const float4*)w)[threadIdx.x];
  __bf16* o = out + row * 1024 + (size_t)threadIdx.x * 4;
  o[0] = (__bf16)(xv.x * rinv * wv.x);
  o[1] = (__bf16)(xv.y * rinv * wv.y);
  o[2] = (__bf16)(xv.z * rinv * wv.z);
  o[3] = (__bf16)(xv.w * rinv * wv.w);
}

// ---------------------------------------------------------------------------
// qkv[BN,3*H*64](bf16) -> rmsnorm(q),rmsnorm(k) transposed to [B,H,D,N](bf16),
// v copied to [B,H,N,D](bf16). One thread per (b,h,n).
// ---------------------------------------------------------------------------
__global__ void __launch_bounds__(256)
qkv_post(const __bf16* __restrict__ qkv, const float* __restrict__ wq,
         const float* __restrict__ wk, __bf16* __restrict__ qt,
         __bf16* __restrict__ kt, __bf16* __restrict__ vv)
{
  const size_t idx = (size_t)blockIdx.x * 256 + threadIdx.x;  // b*65536+h*4096+n
  const int    n   = (int)(idx & 4095);
  const size_t bh  = idx >> 12;
  const size_t bn  = (idx >> 16) * 4096 + n;
  const int    h   = (int)(bh & 15);
  const __bf16* src = qkv + bn * 3072 + (size_t)h * 64;

  float buf[64];
  {  // q -> rmsnorm -> qt[bh][d][n]
    float ss = 0.f;
    for (int d = 0; d < 64; ++d) { float f = (float)src[d]; buf[d] = f; ss += f * f; }
    const float rinv = rsqrtf(ss * (1.0f / 64.0f) + 1e-5f);
    __bf16* dst = qt + bh * (size_t)(64 * 4096) + n;
    for (int d = 0; d < 64; ++d) dst[(size_t)d * 4096] = (__bf16)(buf[d] * rinv * wq[d]);
  }
  {  // k
    const __bf16* s2 = src + 1024;
    float ss = 0.f;
    for (int d = 0; d < 64; ++d) { float f = (float)s2[d]; buf[d] = f; ss += f * f; }
    const float rinv = rsqrtf(ss * (1.0f / 64.0f) + 1e-5f);
    __bf16* dst = kt + bh * (size_t)(64 * 4096) + n;
    for (int d = 0; d < 64; ++d) dst[(size_t)d * 4096] = (__bf16)(buf[d] * rinv * wk[d]);
  }
  {  // v straight copy (already bf16): [bh][n][0..63]
    const uint4* s4 = (const uint4*)(src + 2048);
    uint4* d4 = (uint4*)(vv + (bh * 4096 + (size_t)n) * 64);
    for (int i = 0; i < 8; ++i) d4[i] = s4[i];
  }
}

// ---------------------------------------------------------------------------
// Channel attention per (b,h): S = q^T k * scale (64x64, K=N=4096), softmax
// rows, O = P @ v^T ([64, 4096]); store with the reference's flat-reshape
// permutation into o_merged[BN, 1024] bf16. 16 waves (512 thr) per block.
// ---------------------------------------------------------------------------
__global__ void __launch_bounds__(512)
attn_channel(const __bf16* __restrict__ qt, const __bf16* __restrict__ kt,
             const __bf16* __restrict__ vv, __bf16* __restrict__ o)
{
  __shared__ float  S[64 * 64];
  __shared__ __bf16 P[64 * 64];

  const int bh = blockIdx.x;
  const int b  = bh >> 4;
  const int h  = bh & 15;
  const __bf16* q_ = qt + (size_t)bh * 64 * 4096;   // [D, N] row-major
  const __bf16* k_ = kt + (size_t)bh * 64 * 4096;   // [D, N] row-major
  const __bf16* v_ = vv + (size_t)bh * 4096 * 64;   // [N, D] row-major

  const int lane = threadIdx.x & 31;
  const int w    = threadIdx.x >> 5;

  {  // Phase 1: S tile (i,j); wave w -> (i=w>>2, j=w&3); reduce over N
    const int i = w >> 2, j = w & 3;
    const __bf16* qa = q_ + (size_t)(i * 16) * 4096;
    const __bf16* kb = k_ + (size_t)(j * 16) * 4096;
    v8f acc = {};
    for (int n0 = 0; n0 < 4096; n0 += 32) {
      v16bf a  = load_frag_a(qa + n0, 4096, lane);
      v16bf bb = load_frag_b(kb + n0, 4096, lane);
      acc = wmma_bf16(a, bb, acc);
    }
    const int colt = lane & 15, rbs = (lane >> 4) << 3;
    for (int r = 0; r < 8; ++r)
      S[(i * 16 + rbs + r) * 64 + j * 16 + colt] = acc[r] * 0.125f;  // *D^-0.5
  }
  __syncthreads();

  if (threadIdx.x < 64) {  // Phase 2: softmax over row (axis=-1)
    const int r = threadIdx.x;
    float m = -1e30f;
    for (int c = 0; c < 64; ++c) m = fmaxf(m, S[r * 64 + c]);
    float sum = 0.f;
    for (int c = 0; c < 64; ++c) { float e = __expf(S[r * 64 + c] - m); S[r * 64 + c] = e; sum += e; }
    const float inv = 1.0f / sum;
    for (int c = 0; c < 64; ++c) P[r * 64 + c] = (__bf16)(S[r * 64 + c] * inv);
  }
  __syncthreads();

  {  // Phase 3: O[d, n] = sum_e P[d,e] * v[n,e]; wave w -> dt=w&3, n-tiles w>>2 + 4t
    const int dt = w & 3;
    v16bf aP[2];
    for (int kk = 0; kk < 2; ++kk) {  // A frags from LDS (loop-invariant)
      const int m  = dt * 16 + (lane & 15);
      const int kb = kk * 32 + ((lane >> 4) << 3);
      FragU u;
      u.q[0] = *(const uint4*)&P[m * 64 + kb];
      u.q[1] = *(const uint4*)&P[m * 64 + kb + 16];
      aP[kk] = u.v;
    }
    const int colt = lane & 15, rbs = (lane >> 4) << 3;
    for (int t = 0; t < 64; ++t) {
      const int n0 = ((w >> 2) + 4 * t) * 16;
      v8f acc = {};
      for (int kk = 0; kk < 2; ++kk) {
        v16bf bb = load_frag_b(v_ + (size_t)n0 * 64 + kk * 32, 64, lane);
        acc = wmma_bf16(aP[kk], bb, acc);
      }
      // permuted store: o[b, h*256 + n/16, (n%16)*64 + d]
      for (int r = 0; r < 8; ++r) {
        const int d = dt * 16 + rbs + r;
        const int n = n0 + colt;
        const size_t row = (size_t)b * 4096 + h * 256 + (n >> 4);
        o[row * 1024 + (size_t)((n & 15) * 64 + d)] = (__bf16)acc[r];
      }
    }
  }
}

// ---------------------------------------------------------------------------
__global__ void __launch_bounds__(256)
cvt_f32_bf16(const float* __restrict__ in, __bf16* __restrict__ out, int n)
{
  const int i = blockIdx.x * 256 + threadIdx.x;
  if (i < n) out[i] = (__bf16)in[i];
}

// --------------------------- workspace layout ------------------------------
constexpr size_t BN = 32768;                      // B*N tokens
constexpr size_t SZ_H    = BN * 1024 * 2;         // h / f (bf16)
constexpr size_t SZ_QKV  = BN * 3072 * 2;         // qkv (bf16); o aliases start
constexpr size_t SZ_HEAD = (size_t)8 * 16 * 64 * 4096 * 2;  // one of qt/kt/v
constexpr size_t SZ_X2   = BN * 1024 * 4;         // f32 residual stream

constexpr size_t OFF_H     = 0;
constexpr size_t OFF_QKV   = OFF_H     + SZ_H;
constexpr size_t OFF_QT    = OFF_QKV   + SZ_QKV;  // act[BN,2752]b16 aliases qt..v
constexpr size_t OFF_KT    = OFF_QT    + SZ_HEAD;
constexpr size_t OFF_V     = OFF_KT    + SZ_HEAD;
constexpr size_t OFF_X2    = OFF_V     + SZ_HEAD;
constexpr size_t OFF_WQKV  = OFF_X2    + SZ_X2;
constexpr size_t OFF_WPROJ = OFF_WQKV  + (size_t)3072 * 1024 * 2;
constexpr size_t OFF_WGATE = OFF_WPROJ + (size_t)1024 * 1024 * 2;
constexpr size_t OFF_WUP   = OFF_WGATE + (size_t)2752 * 1024 * 2;
constexpr size_t OFF_WDOWN = OFF_WUP   + (size_t)2752 * 1024 * 2;

extern "C" void kernel_launch(void* const* d_in, const int* in_sizes, int n_in,
                              void* d_out, int out_size, void* d_ws, size_t ws_size,
                              hipStream_t stream)
{
  (void)in_sizes; (void)n_in; (void)out_size; (void)ws_size;
  const float* x      = (const float*)d_in[0];
  const float* w_norm = (const float*)d_in[1];
  const float* w_qkv  = (const float*)d_in[2];
  const float* w_qn   = (const float*)d_in[3];
  const float* w_kn   = (const float*)d_in[4];
  const float* w_proj = (const float*)d_in[5];
  const float* b_proj = (const float*)d_in[6];
  const float* w_nffn = (const float*)d_in[7];
  const float* w_gate = (const float*)d_in[8];
  const float* w_up   = (const float*)d_in[9];
  const float* w_down = (const float*)d_in[10];
  float* out = (float*)d_out;

  char* ws = (char*)d_ws;
  __bf16* h_bf    = (__bf16*)(ws + OFF_H);      // also FFN-norm output 'f'
  __bf16* qkv_bf  = (__bf16*)(ws + OFF_QKV);
  __bf16* o_bf    = (__bf16*)(ws + OFF_QKV);    // alias: qkv dead after qkv_post
  __bf16* qt      = (__bf16*)(ws + OFF_QT);
  __bf16* kt      = (__bf16*)(ws + OFF_KT);
  __bf16* vv      = (__bf16*)(ws + OFF_V);
  __bf16* act     = (__bf16*)(ws + OFF_QT);     // alias: q/k/v dead after attn
  float*  x2      = (float*)(ws + OFF_X2);
  __bf16* wqkv_b  = (__bf16*)(ws + OFF_WQKV);
  __bf16* wproj_b = (__bf16*)(ws + OFF_WPROJ);
  __bf16* wgate_b = (__bf16*)(ws + OFF_WGATE);
  __bf16* wup_b   = (__bf16*)(ws + OFF_WUP);
  __bf16* wdown_b = (__bf16*)(ws + OFF_WDOWN);

  // Weights -> bf16 (deterministic, every call)
  cvt_f32_bf16<<<(3072 * 1024 + 255) / 256, 256, 0, stream>>>(w_qkv,  wqkv_b,  3072 * 1024);
  cvt_f32_bf16<<<(1024 * 1024 + 255) / 256, 256, 0, stream>>>(w_proj, wproj_b, 1024 * 1024);
  cvt_f32_bf16<<<(2752 * 1024 + 255) / 256, 256, 0, stream>>>(w_gate, wgate_b, 2752 * 1024);
  cvt_f32_bf16<<<(2752 * 1024 + 255) / 256, 256, 0, stream>>>(w_up,   wup_b,   2752 * 1024);
  cvt_f32_bf16<<<(1024 * 2752 + 255) / 256, 256, 0, stream>>>(w_down, wdown_b, 1024 * 2752);

  // 1. h = rmsnorm(x)
  rmsnorm_rows<<<32768, 256, 0, stream>>>(x, w_norm, h_bf);
  // 2. qkv = h @ w_qkv^T            [32768, 3072]
  gemm_bf16<0><<<dim3(48, 128), 256, 0, stream>>>(h_bf, wqkv_b, nullptr,
                                                  nullptr, qkv_bf, nullptr, nullptr,
                                                  3072, 1024);
  // 3. split + rmsnorm q,k (transposed) and v
  qkv_post<<<2048, 256, 0, stream>>>(qkv_bf, w_qn, w_kn, qt, kt, vv);
  // 4. channel attention per (b,h), permuted store into o_bf
  attn_channel<<<128, 512, 0, stream>>>(qt, kt, vv, o_bf);
  // 5. x2 = o @ w_proj^T + b_proj + x
  gemm_bf16<1><<<dim3(16, 128), 256, 0, stream>>>(o_bf, wproj_b, nullptr,
                                                  x2, nullptr, b_proj, x,
                                                  1024, 1024);
  // 6. f = rmsnorm(x2)
  rmsnorm_rows<<<32768, 256, 0, stream>>>(x2, w_nffn, h_bf);
  // 7. act = silu(f @ w_gate^T) * (f @ w_up^T)   [32768, 2752]
  gemm_bf16<2><<<dim3(43, 128), 256, 0, stream>>>(h_bf, wgate_b, wup_b,
                                                  nullptr, act, nullptr, nullptr,
                                                  2752, 1024);
  // 8. out = act @ w_down^T + x2
  gemm_bf16<3><<<dim3(16, 128), 256, 0, stream>>>(act, wdown_b, nullptr,
                                                  out, nullptr, nullptr, x2,
                                                  1024, 2752);
}
